// SelfSupervisedMemory_79508434584018
// MI455X (gfx1250) — 8-bit matrix operand order repaired and checked by reading, compiled, not run on hardware
//
#include <hip/hip_runtime.h>
#include <math.h>

typedef __attribute__((ext_vector_type(8))) int   v8i;
typedef __attribute__((ext_vector_type(8))) float v8f;
typedef __attribute__((ext_vector_type(4))) float v4f;
typedef __attribute__((ext_vector_type(2))) float v2f;

#define BS 8
#define N_SLOTS 24
#define NUM_OBJ 20
#define EMBED 128
#define NHEADS 4
#define DHEAD 128
#define PIX (512*512)

#define WS_Q_OFF 8192
#define WS_K_OFF (8192 + BS*24*512)

#define OUT_OBJ 20480
#define OUT_MEMF 40960
#define OUT_TABLE 61440
#define OUT_ATTN 61600

__global__ void init_inter_kernel(int* inter) {
    int idx = blockIdx.x * blockDim.x + threadIdx.x;
    if (idx < BS * 1024) inter[idx] = 0;
}

#define BLOCKS_PER_BATCH 64
#define CHUNK 512
#define ITERS (PIX / (BLOCKS_PER_BATCH * CHUNK))
#define ROWB 528

__global__ void mask_inter_kernel(const float* __restrict__ masks, int* __restrict__ inter) {
    __shared__ unsigned char smem[32 * ROWB];
    __shared__ int interLds[32 * 32];

    const int tid  = threadIdx.x;
    const int b    = blockIdx.x / BLOCKS_PER_BATCH;
    const int blk  = blockIdx.x % BLOCKS_PER_BATCH;
    const int lane = tid & 31;
    const int wave = tid >> 5;
    const int half = (lane >> 4) & 1;
    const int l15  = lane & 15;

    for (int i = tid; i < 1024; i += 256) interLds[i] = 0;
    for (int i = tid; i < (8 * ROWB) / 4; i += 256)
        *(unsigned*)&smem[24 * ROWB + i * 4] = 0;
    __syncthreads();

    v8i acc0 = {}, acc1 = {}, acc2 = {}, acc3 = {};
    const float* mb = masks + (size_t)b * N_SLOTS * PIX;

    for (int it = 0; it < ITERS; ++it) {
        const int p0 = (blk * ITERS + it) * CHUNK;
        for (int i = 0; i < (N_SLOTS * CHUNK) / (4 * 256); ++i) {
            int idx = i * 256 + tid;
            int s = idx >> 7, c4 = idx & 127;
            const v4f* src = (const v4f*)(mb + (size_t)s * PIX + p0);
            v4f v = __builtin_nontemporal_load(&src[c4]);
            unsigned u = (v[0] > 0.5f ? 1u : 0u)
                       | ((v[1] > 0.5f ? 1u : 0u) << 8)
                       | ((v[2] > 0.5f ? 1u : 0u) << 16)
                       | ((v[3] > 0.5f ? 1u : 0u) << 24);
            *(unsigned*)&smem[s * ROWB + c4 * 4] = u;
        }
        __syncthreads();

        const int c0 = wave * 64;
        v8i A0, A1, B0, B1;
        for (int j = 0; j < 8; ++j) {
            int kbA = ((j >> 2) << 5) + (((j >> 1) & 1) << 4) + ((j & 1) << 2) + (half << 3);
            A0[j] = *(const int*)&smem[(0 * 16 + l15) * ROWB + c0 + kbA];
            A1[j] = *(const int*)&smem[(1 * 16 + l15) * ROWB + c0 + kbA];
            int kbB = ((j >> 2) << 5) + (((j >> 1) & 1) << 4) + ((j & 1) << 2) + (half << 3);
            B0[j] = *(const int*)&smem[(0 * 16 + l15) * ROWB + c0 + kbB];
            B1[j] = *(const int*)&smem[(1 * 16 + l15) * ROWB + c0 + kbB];
        }
        acc0 = __builtin_amdgcn_wmma_i32_16x16x64_iu8(false, A0, false, B0, acc0, false, false);
        acc1 = __builtin_amdgcn_wmma_i32_16x16x64_iu8(false, A0, false, B1, acc1, false, false);
        acc2 = __builtin_amdgcn_wmma_i32_16x16x64_iu8(false, A1, false, B0, acc2, false, false);
        acc3 = __builtin_amdgcn_wmma_i32_16x16x64_iu8(false, A1, false, B1, acc3, false, false);
        __syncthreads();
    }

    for (int r = 0; r < 8; ++r) {
        int Mr = r + half * 8;
        atomicAdd(&interLds[(Mr)      * 32 + l15],      acc0[r]);
        atomicAdd(&interLds[(Mr)      * 32 + 16 + l15], acc1[r]);
        atomicAdd(&interLds[(16 + Mr) * 32 + l15],      acc2[r]);
        atomicAdd(&interLds[(16 + Mr) * 32 + 16 + l15], acc3[r]);
    }
    __syncthreads();
    for (int i = tid; i < 1024; i += 256) {
        int v = interLds[i];
        if (v) atomicAdd(&inter[b * 1024 + i], v);
    }
}

__global__ void finalize_kernel(const int* __restrict__ inter,
                                const float* __restrict__ cond,
                                float* __restrict__ out) {
    const int b = blockIdx.x;
    const int t = threadIdx.x;
    __shared__ int msum[N_SLOTS];
    __shared__ int validS[N_SLOTS];
    __shared__ int srcRow[NUM_OBJ];
    __shared__ int numObj;
    const int* I = inter + b * 1024;

    if (t < N_SLOTS) msum[t] = I[t * 32 + t];
    __syncthreads();
    if (t < N_SLOTS) {
        bool bg = (msum[t] == msum[0]);
        bool empty = (msum[t] <= 10);
        bool dup = false;
        for (int j = t + 1; j < N_SLOTS; ++j) {
            float in_ = (float)I[t * 32 + j];
            float un = (float)msum[t] + (float)msum[j] - in_;
            float iou = (un == 0.0f) ? 1.0f : (in_ / fmaxf(un, 1.0f));
            if (iou > 0.5f) dup = true;
        }
        validS[t] = (!(bg || empty || dup)) ? 1 : 0;
    }
    __syncthreads();
    if (t == 0) {
        bool allEmpty = true;
        for (int i = 0; i < N_SLOTS; ++i) if (msum[i] > 10) allEmpty = false;
        if (allEmpty) validS[0] = 1;
        for (int r = 0; r < NUM_OBJ; ++r) srcRow[r] = -1;
        int rank = 0;
        for (int s = 0; s < N_SLOTS; ++s) {
            if (validS[s]) { int d = rank + 1; if (d < NUM_OBJ) srcRow[d] = s; rank++; }
        }
        srcRow[0] = 0;
        numObj = rank;
    }
    __syncthreads();
    for (int r = 0; r < NUM_OBJ; ++r) {
        int s = srcRow[r];
        float v = (s >= 0) ? cond[((size_t)b * N_SLOTS + s) * EMBED + t] : 0.0f;
        out[(size_t)(b * NUM_OBJ + r) * EMBED + t] = v;
    }
    if (t < NUM_OBJ) out[OUT_TABLE + b * NUM_OBJ + t] = (t < numObj + 1) ? 1.0f : 0.0f;
}

__global__ void proj_kernel(const float* __restrict__ obs, const float* __restrict__ pred,
                            const float* __restrict__ wq,  const float* __restrict__ wk,
                            float* __restrict__ Qws, float* __restrict__ Kws) {
    __shared__ float xn[EMBED];
    __shared__ float red[EMBED];
    const int t = threadIdx.x;
    const int id = blockIdx.x;
    const float* x; const float* W; float* outp;
    if (id < BS * 24) {
        x = obs + (size_t)id * EMBED; W = wq; outp = Qws + (size_t)id * 512;
    } else {
        int id2 = id - BS * 24;
        x = pred + (size_t)id2 * EMBED; W = wk; outp = Kws + (size_t)id2 * 512;
    }
    float v = x[t];
    red[t] = v * v;
    __syncthreads();
    for (int s = 64; s > 0; s >>= 1) { if (t < s) red[t] += red[t + s]; __syncthreads(); }
    float inv = 1.0f / fmaxf(sqrtf(red[0]), 1e-12f);
    xn[t] = v * inv;
    __syncthreads();
    const int c0 = t * 4;
    v4f acc = {0.0f, 0.0f, 0.0f, 0.0f};
    for (int d = 0; d < EMBED; ++d) {
        v4f w = *(const v4f*)&W[d * 512 + c0];
        acc += xn[d] * w;
    }
    *(v4f*)&outp[c0] = acc;
}

__global__ void logits_sim_kernel(const float* __restrict__ Qws, const float* __restrict__ Kws,
                                  const float* __restrict__ obs, const float* __restrict__ pred,
                                  float* __restrict__ out) {
    __shared__ float lg[NHEADS * 32 * 32];
    __shared__ float sim[24 * NUM_OBJ];
    __shared__ int   amax[NUM_OBJ];
    __shared__ float pinv[NUM_OBJ];

    const int b = blockIdx.x;
    const int tid = threadIdx.x;
    const int lane = tid & 31, wave = tid >> 5;
    const int half = lane >> 4, l15 = lane & 15;

    for (int jb = wave; jb < 16; jb += 8) {
        int h = jb >> 2, mt = (jb >> 1) & 1, nt = jb & 1;
        int m = mt * 16 + l15;
        int n = nt * 16 + l15;
        bool mval = (m < 24), nval = (n < NUM_OBJ);
        const float* qrow = Qws + (size_t)(b * 24 + m) * 512 + h * DHEAD;
        const float* krow = Kws + (size_t)(b * NUM_OBJ + n) * 512 + h * DHEAD;
        v8f acc = {};
        for (int k0 = 0; k0 < DHEAD; k0 += 4) {
            int kk = k0 + 2 * half;
            v2f a, bb;
            a[0]  = mval ? qrow[kk]     : 0.0f;
            a[1]  = mval ? qrow[kk + 1] : 0.0f;
            bb[0] = nval ? krow[kk]     : 0.0f;
            bb[1] = nval ? krow[kk + 1] : 0.0f;
            acc = __builtin_amdgcn_wmma_f32_16x16x4_f32(false, a, false, bb,
                                                        (short)0, acc, false, false);
        }
        const float scale = 0.0883883476483184f;
        for (int r = 0; r < 8; ++r) {
            int M = mt * 16 + r + half * 8;
            int N = nt * 16 + l15;
            lg[h * 1024 + M * 32 + N] = acc[r] * scale;
        }
    }
    __syncthreads();

    if (tid < NHEADS * 24) {
        int h = tid / 24, q = tid % 24;
        float* row = &lg[h * 1024 + q * 32];
        float mx = row[0];
        for (int k = 1; k < NUM_OBJ; ++k) mx = fmaxf(mx, row[k]);
        float s = 0.0f;
        for (int k = 0; k < NUM_OBJ; ++k) { float e = expf(row[k] - mx); row[k] = e; s += e; }
        float inv = 1.0f / s;
        for (int k = 0; k < NUM_OBJ; ++k) row[k] *= inv;
    }
    __syncthreads();

    for (int e = tid; e < 24 * NUM_OBJ; e += 256) {
        int q = e / NUM_OBJ, k = e % NUM_OBJ;
        float s = 0.25f * (lg[q * 32 + k] + lg[1024 + q * 32 + k] +
                           lg[2048 + q * 32 + k] + lg[3072 + q * 32 + k]);
        sim[e] = s;
        out[OUT_ATTN + (b * 24 + q) * NUM_OBJ + k] = s;
    }
    __syncthreads();

    if (tid < NUM_OBJ) {
        float best = sim[tid]; int arg = 0;
        for (int q = 1; q < 24; ++q) {
            float v = sim[q * NUM_OBJ + tid];
            if (v > best) { best = v; arg = q; }
        }
        amax[tid] = arg;
        const float* pr = pred + (size_t)(b * NUM_OBJ + tid) * EMBED;
        float ss = 0.0f;
        for (int d = 0; d < EMBED; ++d) ss += pr[d] * pr[d];
        pinv[tid] = 1.0f / fmaxf(sqrtf(ss), 1e-12f);
    }
    __syncthreads();

    for (int e = tid; e < NUM_OBJ * EMBED; e += 256) {
        int k = e >> 7, d = e & 127;
        out[OUT_OBJ  + (size_t)(b * NUM_OBJ + k) * EMBED + d] =
            obs[(size_t)(b * 24 + amax[k]) * EMBED + d];
        out[OUT_MEMF + (size_t)(b * NUM_OBJ + k) * EMBED + d] =
            pred[(size_t)(b * NUM_OBJ + k) * EMBED + d] * pinv[k];
    }
}

extern "C" void kernel_launch(void* const* d_in, const int* in_sizes, int n_in,
                              void* d_out, int out_size, void* d_ws, size_t ws_size,
                              hipStream_t stream) {
    const float* conditions = (const float*)d_in[0];
    const float* masks      = (const float*)d_in[1];
    const float* obs        = (const float*)d_in[2];
    const float* pred       = (const float*)d_in[3];
    const float* wq         = (const float*)d_in[4];
    const float* wk         = (const float*)d_in[5];
    float* out = (float*)d_out;

    int*   inter = (int*)d_ws;
    float* Qws   = (float*)d_ws + WS_Q_OFF;
    float* Kws   = (float*)d_ws + WS_K_OFF;

    init_inter_kernel<<<32, 256, 0, stream>>>(inter);
    mask_inter_kernel<<<BS * BLOCKS_PER_BATCH, 256, 0, stream>>>(masks, inter);
    finalize_kernel<<<BS, 128, 0, stream>>>(inter, conditions, out);
    proj_kernel<<<BS * 24 + BS * NUM_OBJ, 128, 0, stream>>>(obs, pred, wq, wk, Qws, Kws);
    logits_sim_kernel<<<BS, 256, 0, stream>>>(Qws, Kws, obs, pred, out);
}
